// Net_18854906430068
// MI455X (gfx1250) — compile-verified, matrix operand order checked by reading, not run on hardware
//
#include <hip/hip_runtime.h>
#include <hip/hip_bf16.h>
#include <math.h>

#ifndef __has_builtin
#define __has_builtin(x) 0
#endif

// ---------------------------------------------------------------------------
// Types for CDNA5 WMMA (wave32). bf16 16x16x32, f32 accumulate.
// ---------------------------------------------------------------------------
typedef __attribute__((ext_vector_type(16))) __bf16 v16bf;
typedef __attribute__((ext_vector_type(8)))  float  v8f;
typedef __attribute__((ext_vector_type(16))) float  v16f;
typedef __attribute__((ext_vector_type(4)))  int    v4i;

#define F0 10
#define F1 10
#define BATCH 4096
#define PDIM 256   // projection width
#define KDIM 512   // fused K for all GEMMs
#define HDIM 128   // final hidden
#define LROW (KDIM + 8)          // padded LDS row (bf16 elems): 1040B stride
#define GEMM_SHMEM (128 * LROW * 2)  // 133,120 B  (320KB/WGP on CDNA5)

#if __has_builtin(__builtin_amdgcn_global_load_async_to_lds_b128)
#define ASYNC_LDS 1
#else
#define ASYNC_LDS 0
#endif

// 16-byte global -> LDS staging chunk (async path; tracked by ASYNCcnt).
__device__ __forceinline__ void stage16(const __bf16* g, __bf16* l) {
#if ASYNC_LDS
    __builtin_amdgcn_global_load_async_to_lds_b128(
        (__attribute__((address_space(1))) v4i*)g,
        (__attribute__((address_space(3))) v4i*)l, 0, 0);
#else
    *(float4*)l = *(const float4*)g;          // global_load_b128 + ds_store_b128
#endif
}

__device__ __forceinline__ void stage_wait() {
#if ASYNC_LDS
 #if __has_builtin(__builtin_amdgcn_s_wait_asynccnt)
    __builtin_amdgcn_s_wait_asynccnt(0);
 #else
    asm volatile("s_wait_asynccnt 0" ::: "memory");
 #endif
#endif
}

// ---------------------------------------------------------------------------
// Weight prep: Wt[n*K + k] = (k < Ka ? Wa[k*N+n] : Wb[(k-Ka)*N+n]) as bf16.
// ---------------------------------------------------------------------------
__global__ void transpose_concat_w(const float* __restrict__ Wa,
                                   const float* __restrict__ Wb,
                                   __bf16* __restrict__ Wt,
                                   int N, int Ka, int Kb) {
    int K = Ka + Kb;
    int i = blockIdx.x * blockDim.x + threadIdx.x;
    if (i >= N * K) return;
    int n = i / K, k = i - n * K;
    float v = (k < Ka) ? Wa[k * N + n] : Wb[(k - Ka) * N + n];
    Wt[i] = (__bf16)v;
}

// ---------------------------------------------------------------------------
// Gather + mean concat:  X[m][0:256]=PA[idx0[m]];  X[m][256:512]=mean PB rows
// ---------------------------------------------------------------------------
__global__ __launch_bounds__(256) void gather_concat(
    const __bf16* __restrict__ PA, const __bf16* __restrict__ PB,
    const int* __restrict__ idx0, const int* __restrict__ idx1,
    __bf16* __restrict__ X, float inv_fan) {
    int m = blockIdx.x;
    int t = threadIdx.x;
    X[(size_t)m * KDIM + t] = PA[(size_t)idx0[m] * PDIM + t];
    float s = 0.f;
    int base = m * F0;
    #pragma unroll
    for (int f = 0; f < F0; ++f)
        s += (float)PB[(size_t)idx1[base + f] * PDIM + t];
    X[(size_t)m * KDIM + PDIM + t] = (__bf16)(s * inv_fan);
}

// X2[m][0:256] = G0[m];  X2[m][256:512] = mean_f G1[m*F0+f]
__global__ __launch_bounds__(256) void concat_self_mean(
    const __bf16* __restrict__ G0, const __bf16* __restrict__ G1,
    __bf16* __restrict__ X2, float inv_fan) {
    int m = blockIdx.x;
    int t = threadIdx.x;
    X2[(size_t)m * KDIM + t] = G0[(size_t)m * PDIM + t];
    float s = 0.f;
    int base = m * F0;
    #pragma unroll
    for (int f = 0; f < F0; ++f)
        s += (float)G1[(size_t)(base + f) * PDIM + t];
    X2[(size_t)m * KDIM + PDIM + t] = (__bf16)(s * inv_fan);
}

// ---------------------------------------------------------------------------
// A-fragment helpers: raw type is fp32 (converted late) or bf16.
// ---------------------------------------------------------------------------
template <bool FP32> struct ARaw;
template <> struct ARaw<true>  { typedef v16f  T; };
template <> struct ARaw<false> { typedef v16bf T; };

__device__ __forceinline__ v16bf cvt_a(v16bf x) { return x; }
__device__ __forceinline__ v16bf cvt_a(v16f x) {
    v16bf r;
    #pragma unroll
    for (int e = 0; e < 16; ++e) r[e] = (__bf16)x[e];
    return r;
}

// ---------------------------------------------------------------------------
// WMMA GEMM: Out[M,N] = act(A[M,512] @ Bt[N,512]^T + bias)
// Block = 256 thr = 8 waves (4 M x 2 N), wave tile 32(M) x 64(N):
//   - B slice [128 x 512] bf16 staged once in LDS (async-to-LDS, padded rows
//     -> conflict-free ds_load_b128 fragment reads)
//   - K-loop unrolled 2x into a ping-pong over two A-fragment buffers, each
//     written only by loads (no loop-carried copies -> no v_mov/v_nop chains),
//     global A always one K-step ahead of its consuming v_wmma.
//   - per K-step: 4 LDS B frags feed 8 v_wmma (each B used by two WMMAs).
// All guards wave-uniform; EXEC all-ones at every WMMA.
// ---------------------------------------------------------------------------
template <bool A_FP32, int ACT, bool OUT_BF16>   // ACT: 0 none, 1 relu, 2 sigmoid
__global__ __launch_bounds__(256) void gemm_wmma(
    const void* __restrict__ Aptr, const __bf16* __restrict__ Bt,
    const float* __restrict__ bias, void* __restrict__ Out,
    int M, int N) {
    constexpr int K = KDIM;
    extern __shared__ __bf16 Bsm[];   // [128][LROW]

    const int tid = threadIdx.x;
    const int nbase = blockIdx.x * 128;   // N is always a multiple of 128 here

    // ---- stage B[128][512] slice into LDS ----
    #pragma unroll 4
    for (int c = tid; c < 128 * (K / 8); c += 256) {   // 8192 16B chunks
        int r  = c >> 6;            // K/8 == 64 chunks per row
        int kc = (c & 63) * 8;
        stage16(Bt + (size_t)(nbase + r) * K + kc, Bsm + r * LROW + kc);
    }
    stage_wait();
    __syncthreads();

    const int lane = tid & 31;
    const int wave = tid >> 5;
    const int mw = wave & 3, nw = wave >> 2;
    const int m0 = blockIdx.y * 128 + mw * 32;
    if (m0 < M) {                                   // uniform per wave
        const int half = lane >> 4;                 // K sub-block select
        const int l15  = lane & 15;
        const int koff = half * 16;
        const bool has2 = (m0 + 16 < M);            // second 16-row sub-tile?
        const int row0 = m0 + l15;                  // valid: M % 16 == 0
        const int row1 = has2 ? (m0 + 16 + l15) : row0;  // clamp when absent

        const __bf16* Ab0 = (const __bf16*)Aptr + (size_t)row0 * K + koff;
        const float*  Af0 = (const float* )Aptr + (size_t)row0 * K + koff;
        const __bf16* Ab1 = (const __bf16*)Aptr + (size_t)row1 * K + koff;
        const float*  Af1 = (const float* )Aptr + (size_t)row1 * K + koff;
        const __bf16* Bl  = Bsm + (nw * 64 + l15) * LROW + koff;

        typedef typename ARaw<A_FP32>::T araw_t;
        auto loadA0 = [&](int k) -> araw_t {
            if constexpr (A_FP32) return *(const v16f*)(Af0 + k);
            else                  return *(const v16bf*)(Ab0 + k);
        };
        auto loadA1 = [&](int k) -> araw_t {
            if constexpr (A_FP32) return *(const v16f*)(Af1 + k);
            else                  return *(const v16bf*)(Ab1 + k);
        };

        v8f c00 = {}, c01 = {}, c02 = {}, c03 = {};   // rows m0..m0+15
        v8f c10 = {}, c11 = {}, c12 = {}, c13 = {};   // rows m0+16..m0+31

        auto do_step = [&](v16bf a0, v16bf a1, int k) {
            const v16bf b0 = *(const v16bf*)(Bl + k);
            const v16bf b1 = *(const v16bf*)(Bl + 16 * LROW + k);
            c00 = __builtin_amdgcn_wmma_f32_16x16x32_bf16(false, a0, false, b0, (short)0, c00, false, false);
            c10 = __builtin_amdgcn_wmma_f32_16x16x32_bf16(false, a1, false, b0, (short)0, c10, false, false);
            const v16bf b2 = *(const v16bf*)(Bl + 32 * LROW + k);
            c01 = __builtin_amdgcn_wmma_f32_16x16x32_bf16(false, a0, false, b1, (short)0, c01, false, false);
            c11 = __builtin_amdgcn_wmma_f32_16x16x32_bf16(false, a1, false, b1, (short)0, c11, false, false);
            const v16bf b3 = *(const v16bf*)(Bl + 48 * LROW + k);
            c02 = __builtin_amdgcn_wmma_f32_16x16x32_bf16(false, a0, false, b2, (short)0, c02, false, false);
            c12 = __builtin_amdgcn_wmma_f32_16x16x32_bf16(false, a1, false, b2, (short)0, c12, false, false);
            c03 = __builtin_amdgcn_wmma_f32_16x16x32_bf16(false, a0, false, b3, (short)0, c03, false, false);
            c13 = __builtin_amdgcn_wmma_f32_16x16x32_bf16(false, a1, false, b3, (short)0, c13, false, false);
        };

        // Ping-pong: pa*/qa* are only ever written by loads (no copies).
        araw_t pa0 = loadA0(0);
        araw_t pa1 = loadA1(0);
        for (int k = 0; k < K; k += 64) {          // K/32 is even
            araw_t qa0 = loadA0(k + 32);           // prefetch odd step
            araw_t qa1 = loadA1(k + 32);
            do_step(cvt_a(pa0), cvt_a(pa1), k);    // compute even step
            if (k + 64 < K) {                      // prefetch next even step
                pa0 = loadA0(k + 64);
                pa1 = loadA1(k + 64);
            }
            do_step(cvt_a(qa0), cvt_a(qa1), k + 32);  // compute odd step
        }

        // Epilogue. D layout: VGPR r -> M = mbase + half*8 + r; lane&15 -> col.
        auto store_tile = [&](v8f acc, int mbase, int j) {
            const int ncol = nbase + nw * 64 + j * 16 + l15;
            const float bv = bias ? bias[ncol] : 0.f;
            #pragma unroll
            for (int r = 0; r < 8; ++r) {
                const int mrow = mbase + half * 8 + r;
                float v = acc[r] + bv;
                if (ACT == 1) v = fmaxf(v, 0.f);
                else if (ACT == 2) v = 1.f / (1.f + __expf(-v));
                if (OUT_BF16) ((__bf16*)Out)[(size_t)mrow * N + ncol] = (__bf16)v;
                else          ((float* )Out)[(size_t)mrow * N + ncol] = v;
            }
        };
        store_tile(c00, m0, 0);
        store_tile(c01, m0, 1);
        store_tile(c02, m0, 2);
        store_tile(c03, m0, 3);
        if (has2) {
            store_tile(c10, m0 + 16, 0);
            store_tile(c11, m0 + 16, 1);
            store_tile(c12, m0 + 16, 2);
            store_tile(c13, m0 + 16, 3);
        }
    }
}

// ---------------------------------------------------------------------------
// logits[b] = sum_d UH[b,d]*IH[b,d]*Wl[d] + bl   (one wave per row)
// ---------------------------------------------------------------------------
__global__ __launch_bounds__(256) void final_logits(
    const float* __restrict__ UH, const float* __restrict__ IH,
    const float* __restrict__ Wl, const float* __restrict__ bl,
    float* __restrict__ out, int B) {
    const int lane = threadIdx.x & 31;
    const int wave = threadIdx.x >> 5;
    const int b = blockIdx.x * 8 + wave;
    if (b >= B) return;
    float s = 0.f;
    #pragma unroll
    for (int j = 0; j < HDIM / 32; ++j) {
        int d = lane + 32 * j;
        s += UH[(size_t)b * HDIM + d] * IH[(size_t)b * HDIM + d] * Wl[d];
    }
    #pragma unroll
    for (int off = 16; off; off >>= 1) s += __shfl_xor(s, off, 32);
    if (lane == 0) out[b] = s + bl[0];
}

// ---------------------------------------------------------------------------
extern "C" void kernel_launch(void* const* d_in, const int* in_sizes, int n_in,
                              void* d_out, int out_size, void* d_ws, size_t ws_size,
                              hipStream_t stream) {
    const float* user_feat = (const float*)d_in[0];   // [50000,512]
    const float* item_feat = (const float*)d_in[1];   // [100000,512]
    const float* W_pu   = (const float*)d_in[2];
    const float* b_pu   = (const float*)d_in[3];
    const float* W_pi   = (const float*)d_in[4];
    const float* b_pi   = (const float*)d_in[5];
    const float* u_self0 = (const float*)d_in[6];
    const float* u_agg0  = (const float*)d_in[7];
    const float* u_self1 = (const float*)d_in[8];
    const float* u_agg1  = (const float*)d_in[9];
    const float* i_self0 = (const float*)d_in[10];
    const float* i_agg0  = (const float*)d_in[11];
    const float* i_self1 = (const float*)d_in[12];
    const float* i_agg1  = (const float*)d_in[13];
    const float* W_lin  = (const float*)d_in[14];
    const float* b_lin  = (const float*)d_in[15];
    const int* src_h0 = (const int*)d_in[16];
    const int* src_h1 = (const int*)d_in[17];
    const int* src_h2 = (const int*)d_in[18];
    const int* dst_h0 = (const int*)d_in[19];
    const int* dst_h1 = (const int*)d_in[20];
    const int* dst_h2 = (const int*)d_in[21];
    float* out = (float*)d_out;

    const int NU = 50000, NI = 100000;

    // ---- workspace carve-up ----
    char* ws = (char*)d_ws;
    size_t off = 0;
    auto carve = [&](size_t bytes) {
        void* p = ws + off;
        off += (bytes + 255) & ~(size_t)255;
        return p;
    };
    __bf16* wt_pu = (__bf16*)carve((size_t)PDIM * KDIM * 2);
    __bf16* wt_pi = (__bf16*)carve((size_t)PDIM * KDIM * 2);
    __bf16* wt_u0 = (__bf16*)carve((size_t)PDIM * KDIM * 2);
    __bf16* wt_i0 = (__bf16*)carve((size_t)PDIM * KDIM * 2);
    __bf16* wt_u1 = (__bf16*)carve((size_t)HDIM * KDIM * 2);
    __bf16* wt_i1 = (__bf16*)carve((size_t)HDIM * KDIM * 2);
    __bf16* u_proj = (__bf16*)carve((size_t)NU * PDIM * 2);   // 25.6 MB
    __bf16* i_proj = (__bf16*)carve((size_t)NI * PDIM * 2);   // 51.2 MB (L2-resident)
    __bf16* X0 = (__bf16*)carve((size_t)BATCH * KDIM * 2);
    __bf16* X1 = (__bf16*)carve((size_t)BATCH * F0 * KDIM * 2);
    __bf16* G0 = (__bf16*)carve((size_t)BATCH * PDIM * 2);
    __bf16* G1 = (__bf16*)carve((size_t)BATCH * F0 * PDIM * 2);
    __bf16* X2 = (__bf16*)carve((size_t)BATCH * KDIM * 2);
    float*  Hu = (float*)carve((size_t)BATCH * HDIM * 4);
    float*  Hi = (float*)carve((size_t)BATCH * HDIM * 4);
    (void)ws_size; (void)in_sizes; (void)n_in; (void)out_size;

    // ---- 1) weight prep ----
    {
        int n1 = PDIM * KDIM, blocks1 = (n1 + 255) / 256;
        int n2 = HDIM * KDIM, blocks2 = (n2 + 255) / 256;
        transpose_concat_w<<<blocks1, 256, 0, stream>>>(W_pu, W_pu, wt_pu, PDIM, KDIM, 0);
        transpose_concat_w<<<blocks1, 256, 0, stream>>>(W_pi, W_pi, wt_pi, PDIM, KDIM, 0);
        transpose_concat_w<<<blocks1, 256, 0, stream>>>(u_self0, u_agg0, wt_u0, PDIM, PDIM, PDIM);
        transpose_concat_w<<<blocks1, 256, 0, stream>>>(i_self0, i_agg0, wt_i0, PDIM, PDIM, PDIM);
        transpose_concat_w<<<blocks2, 256, 0, stream>>>(u_self1, u_agg1, wt_u1, HDIM, PDIM, PDIM);
        transpose_concat_w<<<blocks2, 256, 0, stream>>>(i_self1, i_agg1, wt_i1, HDIM, PDIM, PDIM);
    }

    // ---- 2) projection GEMMs (fp32 A -> bf16 WMMA -> sigmoid -> bf16) ----
    gemm_wmma<true, 2, true><<<dim3(PDIM / 128, (NU + 127) / 128), 256, GEMM_SHMEM, stream>>>(
        user_feat, wt_pu, b_pu, u_proj, NU, PDIM);
    gemm_wmma<true, 2, true><<<dim3(PDIM / 128, (NI + 127) / 128), 256, GEMM_SHMEM, stream>>>(
        item_feat, wt_pi, b_pi, i_proj, NI, PDIM);

    // ---- 3) src (user) side ----
    gather_concat<<<BATCH, 256, 0, stream>>>(u_proj, i_proj, src_h0, src_h1, X0, 1.f / F0);
    gather_concat<<<BATCH * F0, 256, 0, stream>>>(i_proj, u_proj, src_h1, src_h2, X1, 1.f / F1);
    gemm_wmma<false, 1, true><<<dim3(PDIM / 128, BATCH / 128), 256, GEMM_SHMEM, stream>>>(
        X0, wt_u0, nullptr, G0, BATCH, PDIM);
    gemm_wmma<false, 1, true><<<dim3(PDIM / 128, BATCH * F0 / 128), 256, GEMM_SHMEM, stream>>>(
        X1, wt_u0, nullptr, G1, BATCH * F0, PDIM);
    concat_self_mean<<<BATCH, 256, 0, stream>>>(G0, G1, X2, 1.f / F0);
    gemm_wmma<false, 0, false><<<dim3(1, BATCH / 128), 256, GEMM_SHMEM, stream>>>(
        X2, wt_u1, nullptr, Hu, BATCH, HDIM);

    // ---- 4) dst (item) side (reuse scratch) ----
    gather_concat<<<BATCH, 256, 0, stream>>>(i_proj, u_proj, dst_h0, dst_h1, X0, 1.f / F0);
    gather_concat<<<BATCH * F0, 256, 0, stream>>>(u_proj, i_proj, dst_h1, dst_h2, X1, 1.f / F1);
    gemm_wmma<false, 1, true><<<dim3(PDIM / 128, BATCH / 128), 256, GEMM_SHMEM, stream>>>(
        X0, wt_i0, nullptr, G0, BATCH, PDIM);
    gemm_wmma<false, 1, true><<<dim3(PDIM / 128, BATCH * F0 / 128), 256, GEMM_SHMEM, stream>>>(
        X1, wt_i0, nullptr, G1, BATCH * F0, PDIM);
    concat_self_mean<<<BATCH, 256, 0, stream>>>(G0, G1, X2, 1.f / F0);
    gemm_wmma<false, 0, false><<<dim3(1, BATCH / 128), 256, GEMM_SHMEM, stream>>>(
        X2, wt_i1, nullptr, Hi, BATCH, HDIM);

    // ---- 5) logits ----
    final_logits<<<(BATCH + 7) / 8, 256, 0, stream>>>(Hu, Hi, W_lin, b_lin, out, BATCH);
}
